// PolarQuant_62328565400264
// MI455X (gfx1250) — matrix operand order repaired and checked by reading, not compiled, not run on hardware
//
#include <hip/hip_runtime.h>
#include <math.h>

// PolarQuant for MI455X (gfx1250, wave32).
// Two 256x256 GEMMs done with v_wmma_f32_16x16x32_bf16 (f32 accumulate),
// quantization done entirely in the cos-domain (no arccos / no searchsorted),
// reconstruction via an 8-entry sin/cos LUT + running product.

#define DIM   256
#define DIMP  260   // LDS row pad (breaks 64-bank conflicts for 16-row strided reads)
#define RPB   32    // rows per block
#define NLEV  8
#define EPSF  1e-8f

typedef __attribute__((ext_vector_type(16))) __bf16 v16bf;
typedef __attribute__((ext_vector_type(8)))  float  v8f;

__device__ __forceinline__ v16bf pack_bf16x16(float4 a0, float4 a1, float4 a2, float4 a3) {
    v16bf r;
    r[0]  = (__bf16)a0.x; r[1]  = (__bf16)a0.y; r[2]  = (__bf16)a0.z; r[3]  = (__bf16)a0.w;
    r[4]  = (__bf16)a1.x; r[5]  = (__bf16)a1.y; r[6]  = (__bf16)a1.z; r[7]  = (__bf16)a1.w;
    r[8]  = (__bf16)a2.x; r[9]  = (__bf16)a2.y; r[10] = (__bf16)a2.z; r[11] = (__bf16)a2.w;
    r[12] = (__bf16)a3.x; r[13] = (__bf16)a3.y; r[14] = (__bf16)a3.z; r[15] = (__bf16)a3.w;
    return r;
}

extern "C" __global__ __launch_bounds__(256)
void polarquant_kernel(const float* __restrict__ X,    // (nrows, 256) already row-major
                       const float* __restrict__ H,    // (256, 256)
                       const float* __restrict__ Dv,   // (256,)
                       const float* __restrict__ Cen,  // (8,)
                       const float* __restrict__ Bnd,  // (10,)
                       float* __restrict__ outR,       // (nrows,)
                       float* __restrict__ outI,       // (nrows, 255) idx as float
                       float* __restrict__ outRec)     // (nrows, 256)
{
    __shared__ float         lX[RPB][DIMP];     // xD -> x_rht -> rec (reused)
    __shared__ unsigned char lIdx[RPB][DIM];    // per-row quant indices

    const int tid     = threadIdx.x;
    const int wave    = tid >> 5;
    const int lane    = tid & 31;
    const int ln15    = lane & 15;
    const int lhalfA  = (lane & 16) ? 8  : 0;   // A-fragment K sub-offset (16-bit A layout)
    const int lhalfB  = (lane & 16) ? 16 : 0;   // K half for contiguous k order (phase 1 A and B)
    const int rowbase = blockIdx.x * RPB;

    // H is L2-resident (256 KB << 192 MB L2); warm it: emits global_prefetch_b8.
    __builtin_prefetch(H + (size_t)tid * DIM, 0, 1);

    // ---------------- Phase 0: stage xD into LDS (coalesced) ----------------
    const float dcol = Dv[tid];
    #pragma unroll 4
    for (int i = 0; i < RPB; ++i)
        lX[i][tid] = X[(size_t)(rowbase + i) * DIM + tid] * dcol;
    __syncthreads();

    // ---------------- Phase 1: x_rht = (xD) @ H^T  via WMMA bf16 -------------
    // Output 32x256 = 2 M-tiles x 16 N-tiles of 16x16; wave w owns tiles 4w..4w+3.
    // B[k][n] = H^T[k][n] = H[n][k]  -> contiguous reads of H row n.
    v8f acc[4];
    #pragma unroll
    for (int t = 0; t < 4; ++t) {
        const int tile = wave * 4 + t;
        const int mt = tile >> 4, nt = tile & 15;
        const float* arow = &lX[mt * 16 + ln15][0];
        const float* brow = H + (size_t)(nt * 16 + ln15) * DIM;
        v8f c = {0.f, 0.f, 0.f, 0.f, 0.f, 0.f, 0.f, 0.f};
        #pragma unroll
        for (int kk = 0; kk < DIM; kk += 32) {
            // A fragment: row M=ln15, K = kk+lhalfB .. +15 (contiguous, matches B)
            float4 a0 = *(const float4*)(arow + kk + lhalfB);
            float4 a1 = *(const float4*)(arow + kk + lhalfB + 4);
            float4 a2 = *(const float4*)(arow + kk + lhalfB + 8);
            float4 a3 = *(const float4*)(arow + kk + lhalfB + 12);
            v16bf af = pack_bf16x16(a0, a1, a2, a3);
            // B fragment: N=ln15, K = kk+lhalfB .. +15 (contiguous in H row)
            float4 b0 = *(const float4*)(brow + kk + lhalfB);
            float4 b1 = *(const float4*)(brow + kk + lhalfB + 4);
            float4 b2 = *(const float4*)(brow + kk + lhalfB + 8);
            float4 b3 = *(const float4*)(brow + kk + lhalfB + 12);
            v16bf bf = pack_bf16x16(b0, b1, b2, b3);
            c = __builtin_amdgcn_wmma_f32_16x16x32_bf16(false, af, false, bf,
                                                        (short)0, c, false, false);
        }
        acc[t] = c;
    }
    __syncthreads();  // all A-fragment reads of lX complete before overwrite
    #pragma unroll
    for (int t = 0; t < 4; ++t) {
        const int tile = wave * 4 + t;
        const int mt = tile >> 4, nt = tile & 15;
        const int col = nt * 16 + ln15;
        const int rb  = mt * 16 + ((lane & 16) ? 8 : 0);
        #pragma unroll
        for (int r = 0; r < 8; ++r) lX[rb + r][col] = acc[t][r];  // C/D layout
    }
    __syncthreads();

    // ---------------- Phase 2: polar quantize (cos-domain, no arccos) --------
    if (tid < RPB) {
        // cos is monotone decreasing on [0,pi]:
        //   theta >= Bnd[k]  <=>  cos(theta) <= cos(Bnd[k])
        float cb[NLEV], cc[NLEV], sc[NLEV];
        #pragma unroll
        for (int k = 0; k < NLEV; ++k) {
            cb[k] = cosf(Bnd[k + 1]);   // interior boundaries 1..8
            cc[k] = cosf(Cen[k]);
            sc[k] = sinf(Cen[k]);
        }
        const int g = rowbase + tid;
        float* p = &lX[tid][0];

        // Reverse scan: suffix sum of squares -> rem_norm, cos_theta, idx.
        float s = 0.f, rsave = 0.f;
        for (int j = DIM - 1; j >= 0; --j) {
            float v = p[j];
            s = fmaf(v, v, s);
            float rn = sqrtf(s) + EPSF;           // rem_norm[j]
            if (j < DIM - 1) {
                float ct = v / rn;
                ct = fminf(fmaxf(ct, -1.f + EPSF), 1.f - EPSF);
                int cnt = 0;
                #pragma unroll
                for (int k = 0; k < NLEV; ++k) cnt += (ct <= cb[k]) ? 1 : 0;
                int q = cnt > (NLEV - 1) ? (NLEV - 1) : cnt;
                lIdx[tid][j] = (unsigned char)q;
                outI[(size_t)g * (DIM - 1) + j] = (float)q;
            }
            if (j == 0) rsave = rn;               // r = rem_norm[0]
        }
        outR[g] = rsave;

        // Forward scan: rec[j] = r * cos(theta_q[j]) * prod_{i<j} sin(theta_q[i])
        float prod = rsave;
        for (int j = 0; j < DIM - 1; ++j) {
            int q = lIdx[tid][j];
            p[j] = prod * cc[q];
            prod *= sc[q];
        }
        p[DIM - 1] = prod;
    }
    __syncthreads();

    // ---------------- Phase 3: x_rec = (rec @ H) * D  via WMMA bf16 ----------
    // B[k][n] = H[k][n]  -> strided column reads of H (L2-resident).
    #pragma unroll
    for (int t = 0; t < 4; ++t) {
        const int tile = wave * 4 + t;
        const int mt = tile >> 4, nt = tile & 15;
        const int n  = nt * 16 + ln15;
        const float* arow = &lX[mt * 16 + ln15][0];
        v8f c = {0.f, 0.f, 0.f, 0.f, 0.f, 0.f, 0.f, 0.f};
        #pragma unroll
        for (int kk = 0; kk < DIM; kk += 32) {
            float4 a0 = *(const float4*)(arow + kk + lhalfA);
            float4 a1 = *(const float4*)(arow + kk + lhalfA + 4);
            float4 a2 = *(const float4*)(arow + kk + 16 + lhalfA);
            float4 a3 = *(const float4*)(arow + kk + 16 + lhalfA + 4);
            v16bf af = pack_bf16x16(a0, a1, a2, a3);
            v16bf bf;
            #pragma unroll
            for (int j = 0; j < 16; ++j)
                bf[j] = (__bf16)H[(size_t)(kk + lhalfA + j + (j >= 8 ? 8 : 0)) * DIM + n];
            c = __builtin_amdgcn_wmma_f32_16x16x32_bf16(false, af, false, bf,
                                                        (short)0, c, false, false);
        }
        const float dn = Dv[n];
        const int rb = mt * 16 + ((lane & 16) ? 8 : 0);
        #pragma unroll
        for (int r = 0; r < 8; ++r)
            outRec[(size_t)(rowbase + rb + r) * DIM + n] = c[r] * dn;
    }
}

extern "C" void kernel_launch(void* const* d_in, const int* in_sizes, int n_in,
                              void* d_out, int out_size, void* d_ws, size_t ws_size,
                              hipStream_t stream) {
    const float* x   = (const float*)d_in[0];   // (8, 8192, 256) f32
    const float* H   = (const float*)d_in[1];   // (256, 256) f32
    const float* D   = (const float*)d_in[2];   // (256,) f32
    const float* cen = (const float*)d_in[3];   // (8,) f32
    const float* bnd = (const float*)d_in[4];   // (10,) f32

    const int nrows = in_sizes[0] / DIM;        // 65536

    // outputs concatenated flat in return order: r | idx | x_rec
    float* out_r   = (float*)d_out;
    float* out_idx = out_r + nrows;
    float* out_rec = out_r + (size_t)nrows * DIM;  // nrows*(1+255)

    dim3 grid(nrows / RPB), block(256);
    hipLaunchKernelGGL(polarquant_kernel, grid, block, 0, stream,
                       x, H, D, cen, bnd, out_r, out_idx, out_rec);
}
